// CrossOp_53781580480707
// MI455X (gfx1250) — matrix operand order repaired and checked by reading, not compiled, not run on hardware
//
#include <hip/hip_runtime.h>

typedef __attribute__((ext_vector_type(16))) __bf16 v16bf;
typedef __attribute__((ext_vector_type(8)))  float  v8f;
typedef unsigned int u32x4 __attribute__((ext_vector_type(4)));
typedef int          i32x4 __attribute__((ext_vector_type(4)));
typedef int          i32x8 __attribute__((ext_vector_type(8)));

#define Bn   4
#define SYn  16
#define Cc   64          // channels per source (target / support)
#define COn  64
#define Hh   128
#define Wn   128
#define ROWS 4           // output rows per workgroup
#define NIMG (Bn * SYn)          // 64 fused (b,sy) images
#define HW   (Hh * Wn)           // 16384
#define NT_ELEMS (Bn * COn * HW) // 4,194,304 floats (new_target)
#define WFRAG_HALFS (9 * 4 * 4 * 512) // 73728 bf16 = 147,456 B packed weights
#define WFRAG_QW    (WFRAG_HALFS / 4) // 18432 8-byte units

__device__ __forceinline__ unsigned short f2bf(float f) {
  union { float f; unsigned u; } v; v.f = f;
  unsigned r = v.u + 0x7FFFu + ((v.u >> 16) & 1u); // round-to-nearest-even
  return (unsigned short)(r >> 16);
}

// ---------------------------------------------------------------------------
// Kernel 1: pack weight[CO=64][2C=128][3][3] f32 -> bf16 A-fragment layout.
// Fragment f = (tap*4 + ciChunk)*4 + coTile ; within: [lane(32)][h(16)] halfs.
// A 16x32 bf16 lane layout used here (contiguous k; B uses the same order):
//   lane<16 : M=lane,    K = h
//   lane>=16: M=lane-16, K = h + 16
// ---------------------------------------------------------------------------
__global__ void pack_weights(const float* __restrict__ w,
                             unsigned short* __restrict__ out) {
  int gidx = blockIdx.x * 256 + threadIdx.x;
  if (gidx >= WFRAG_HALFS) return;
  int f    = gidx >> 9;          // /512
  int rr   = gidx & 511;
  int lane = rr >> 4;
  int h    = rr & 15;
  int ct   = f & 3;              // co tile
  int qg   = (f >> 2) & 3;       // ci chunk of 32
  int t    = f >> 4;             // tap 0..8
  int kh   = t / 3, kw = t - kh * 3;
  int m    = lane & 15;
  int k    = h + ((lane >> 4) << 4);
  int ci   = qg * 32 + k;
  int co   = ct * 16 + m;
  out[gidx] = f2bf(w[((co * 128 + ci) * 3 + kh) * 3 + kw]);
}

// ---------------------------------------------------------------------------
// Kernel 2: implicit-GEMM 3x3 conv via v_wmma_f32_16x16x32_bf16.
// Grid: (y-blocks=32, n=64). Block: 256 threads = 8 waves.
// TDM (tensor_load_to_lds) DMAs all 144 KB of packed bf16 weight fragments
// into LDS once at workgroup start (TENSORcnt path); inputs are staged with
// float4 reads + bf16 transpose-scatter; bias arrives via async global->LDS.
// ---------------------------------------------------------------------------
__global__ __launch_bounds__(256)
void conv_wmma(const float* __restrict__ target,
               const float* __restrict__ support,
               const unsigned short* __restrict__ wfrag,
               const float* __restrict__ bias,
               float* __restrict__ inter) {
  __shared__ __align__(16) unsigned short lds_in[ROWS + 2][130][64]; //  99,840 B
  __shared__ __align__(16) unsigned short lds_w[WFRAG_HALFS];        // 147,456 B
  __shared__ float lds_bias[COn];

  const int y0   = blockIdx.x * ROWS;
  const int n    = blockIdx.y;       // b*16 + sy
  const int b    = n >> 4;
  const int tid  = threadIdx.x;
  const int lane = tid & 31;
  const int wid  = tid >> 5;
  const int ct   = wid & 3;          // co tile
  const int ph   = wid >> 2;         // pixel half (0/1)
  const int lhi  = lane >> 4;        // 0/1
  const int llo  = lane & 15;

  // ---- CDNA5 TDM: DMA the whole packed-weight block (144 KB) into LDS ----
#if defined(__has_builtin) && __has_builtin(__builtin_amdgcn_tensor_load_to_lds)
  if (wid == 0) {
    unsigned ldsa = (unsigned)(size_t)&lds_w[0];        // flat low32 = LDS offset
    unsigned long long ga = (unsigned long long)(size_t)wfrag;
    // D# group0: count=1 | lds_addr | global_addr[56:0] | type=2
    u32x4 g0 = { 1u, ldsa, (unsigned)ga,
                 (unsigned)((ga >> 32) & 0x01FFFFFFu) | (2u << 30) };
    // D# group1: data_size=3 (8B units); tensor_dim0=18432; tensor_dim1=1;
    // tile_dim0=18432; tile_dim1=1; tile_dim2=0; dim0_stride=18432.
    i32x8 g1 = { (int)(3u << 16),
                 (int)((WFRAG_QW & 0xFFFF) << 16),      // tensor_dim0[15:0]<<16
                 (int)((WFRAG_QW >> 16) | (1 << 16)),   // td0[31:16] | td1.lo<<16
                 (int)((WFRAG_QW & 0xFFFF) << 16),      // td1.hi | tile_dim0<<16
                 1,                                     // tile_dim1=1, tile_dim2=0
                 WFRAG_QW,                              // dim0_stride[31:0]
                 0, 0 };
    i32x4 gz  = { 0, 0, 0, 0 };
#if __clang_major__ >= 23
    i32x8 gz8 = { 0, 0, 0, 0, 0, 0, 0, 0 };
    __builtin_amdgcn_tensor_load_to_lds(g0, g1, gz, gz, gz8, 0);
#else
    __builtin_amdgcn_tensor_load_to_lds(g0, g1, gz, gz, 0);
#endif
  }
#define TDM_WAIT() do {                                           \
    if (wid == 0) {                                               \
      __builtin_amdgcn_s_wait_tensorcnt(0);                       \
    } } while (0)
#else
  // Fallback: async global->LDS copy loop (validated mnemonic), 16B per lane.
  for (int c = tid; c < WFRAG_HALFS / 8; c += 256) {
    unsigned ldsa = (unsigned)(size_t)&lds_w[c * 8];
    const unsigned short* g = wfrag + c * 8;
    asm volatile("global_load_async_to_lds_b128 %0, %1, off"
                 :: "v"(ldsa), "v"(g) : "memory");
  }
#define TDM_WAIT() asm volatile("s_wait_asynccnt 0" ::: "memory")
#endif

  // ---- CDNA5 async global->LDS (ASYNCcnt path): stage the bias vector ----
  if (tid < COn) {
    unsigned ldsoff = (unsigned)(size_t)&lds_bias[tid];
    const float* g  = bias + tid;
    asm volatile("global_load_async_to_lds_b32 %0, %1, off"
                 :: "v"(ldsoff), "v"(g) : "memory");
  }

  const v8f zero = {0.f, 0.f, 0.f, 0.f, 0.f, 0.f, 0.f, 0.f};
  v8f acc[ROWS * 4];
  #pragma unroll
  for (int i = 0; i < ROWS * 4; ++i) acc[i] = zero;

  const float* srcT = target  + (long)b * (Cc * HW);
  const float* srcS = support + (long)n * (Cc * HW);

  for (int ch = 0; ch < 2; ++ch) {
    const float* src = (ch == 0) ? srcT : srcS;
    __syncthreads(); // previous stage's readers done before overwrite

    // Interior staging: float4 (b128) coalesced reads along w, bf16
    // transpose-scatter into LDS. Rows y0-1 .. y0+4; OOB rows -> zeros.
    for (int idx = tid; idx < (ROWS + 2) * 64 * 32; idx += 256) {
      int r   = idx / (64 * 32);
      int rem = idx - r * (64 * 32);
      int ci  = rem >> 5;
      int xx  = (rem & 31) << 2;     // 0..124, step 4
      int yy  = y0 + r - 1;
      float4 v = make_float4(0.f, 0.f, 0.f, 0.f);
      if ((unsigned)yy < (unsigned)Hh) {
        long go = ((long)ci * Hh + yy) * Wn + xx;
        v = *(const float4*)(src + go);
        if (ch == 0) __builtin_prefetch(srcS + go, 0, 0); // overlap next stage
      }
      unsigned short* d = &lds_in[r][xx + 1][ci];
      d[0]   = f2bf(v.x);
      d[64]  = f2bf(v.y);
      d[128] = f2bf(v.z);
      d[192] = f2bf(v.w);
    }
    // Halo columns (x = -1 and x = 128) -> zeros.
    for (int i = tid; i < (ROWS + 2) * 64 * 2; i += 256) {
      int r    = i >> 7;
      int rem  = i & 127;
      int ci   = rem & 63;
      int side = rem >> 6;
      lds_in[r][side ? 129 : 0][ci] = 0;
    }
    if (ch == 0) TDM_WAIT(); // weights must be resident before first use
    __syncthreads();

    for (int t = 0; t < 9; ++t) {
      const int kh = t / 3;
      const int kw = t - kh * 3;
      #pragma unroll
      for (int q = 0; q < 2; ++q) {
        const int fq = (t * 4 + ch * 2 + q) * 4 + ct;
        v16bf a = *(const v16bf*)&lds_w[fq * 512 + lane * 16]; // LDS A-frag
        const int cio = q * 32 + lhi * 16;
        #pragma unroll
        for (int pt = 0; pt < 4; ++pt) {
          const int pxs = ph * 64 + pt * 16 + llo + kw;
          #pragma unroll
          for (int ry = 0; ry < ROWS; ++ry) {
            v16bf bf = *(const v16bf*)&lds_in[ry + kh][pxs][cio];
            acc[ry * 4 + pt] = __builtin_amdgcn_wmma_f32_16x16x32_bf16(
                false, a, false, bf, (short)0, acc[ry * 4 + pt], false, false);
          }
        }
      }
    }
  }

  // Ensure async bias copy landed in LDS, then epilogue.
#if __has_builtin(__builtin_amdgcn_s_wait_asynccnt)
  __builtin_amdgcn_s_wait_asynccnt(0);
#else
  asm volatile("s_wait_asynccnt 0" ::: "memory");
#endif
  __syncthreads();

  float bs[8];
  #pragma unroll
  for (int r = 0; r < 8; ++r) bs[r] = lds_bias[ct * 16 + lhi * 8 + r];

  // C/D layout: lane<16 -> M=r ; lane>=16 -> M=r+8.
  #pragma unroll
  for (int ry = 0; ry < ROWS; ++ry) {
    float* obase = inter + (long)n * COn * HW + (long)(y0 + ry) * Wn;
    #pragma unroll
    for (int pt = 0; pt < 4; ++pt) {
      const int px = ph * 64 + pt * 16 + llo;
      #pragma unroll
      for (int r = 0; r < 8; ++r) {
        const int co = ct * 16 + lhi * 8 + r;
        obase[(long)co * HW + px] = acc[ry * 4 + pt][r] + bs[r];
      }
    }
  }
}

// ---------------------------------------------------------------------------
// Kernel 3: new_target[b,0,co,h,w] = mean over sy of interaction.
// ---------------------------------------------------------------------------
__global__ void mean_over_sy(const float* __restrict__ inter,
                             float* __restrict__ nt) {
  int idx = blockIdx.x * 256 + threadIdx.x;
  if (idx >= NT_ELEMS) return;
  int w_ = idx & 127;
  int h_ = (idx >> 7) & 127;
  int co = (idx >> 14) & 63;
  int b  = idx >> 20;
  long base = ((long)b * SYn * COn + co) * HW + h_ * Wn + w_;
  float s = 0.f;
  #pragma unroll
  for (int sy = 0; sy < SYn; ++sy) s += inter[base + (long)sy * COn * HW];
  nt[idx] = s * (1.0f / SYn);
}

// ---------------------------------------------------------------------------
extern "C" void kernel_launch(void* const* d_in, const int* in_sizes, int n_in,
                              void* d_out, int out_size, void* d_ws, size_t ws_size,
                              hipStream_t stream) {
  const float* target  = (const float*)d_in[0];
  const float* support = (const float*)d_in[1];
  const float* weight  = (const float*)d_in[2];
  const float* bias    = (const float*)d_in[3];

  float* out        = (float*)d_out;
  float* new_target = out;             // [4,1,64,128,128]
  float* inter      = out + NT_ELEMS;  // [4,16,64,128,128]
  unsigned short* wfrag = (unsigned short*)d_ws; // 144 KB packed bf16 weights

  pack_weights<<<(WFRAG_HALFS + 255) / 256, 256, 0, stream>>>(weight, wfrag);
  conv_wmma<<<dim3(Hh / ROWS, NIMG), 256, 0, stream>>>(target, support, wfrag,
                                                       bias, inter);
  mean_over_sy<<<(NT_ELEMS + 255) / 256, 256, 0, stream>>>(inter, new_target);
}
